// ConvAttentionBlock_48318382080312
// MI455X (gfx1250) — matrix operand order repaired and checked by reading, not compiled, not run on hardware
//
#include <hip/hip_runtime.h>
#include <hip/hip_bf16.h>

typedef __attribute__((ext_vector_type(16))) _Float16 v16h;
typedef __attribute__((ext_vector_type(8)))  _Float16 v8h;
typedef __attribute__((ext_vector_type(8)))  float    v8f;
typedef __attribute__((ext_vector_type(4)))  float    v4f;
typedef __attribute__((ext_vector_type(4)))  unsigned int u32x4;
typedef __attribute__((ext_vector_type(8)))  int      i32x8;
typedef __attribute__((ext_vector_type(4)))  int      i32x4;

#define WMMA_F16(a, b, c) \
    __builtin_amdgcn_wmma_f32_16x16x32_f16(false, (a), false, (b), (short)0, (c), false, false)

constexpr int Bn = 8, Hh = 64, Ww = 64, Cc = 128, HCc = 128;
constexpr int Nn = Hh * Ww;  // 4096

#if __has_builtin(__builtin_amdgcn_tensor_load_to_lds) && \
    __has_builtin(__builtin_amdgcn_s_wait_tensorcnt)
#define USE_TDM 1
#else
#define USE_TDM 0
#endif

// ---------------- fragment loaders (wave32, 16x16x32 f16) ----------------
// A-frag (16x32, lane->M=l%16): elem j -> K = 16h+j
__device__ __forceinline__ v16h load_a_h(const _Float16* row, int h) {
    v8h lo = *(const v8h*)(row + 16 * h);
    v8h hi = *(const v8h*)(row + 16 * h + 8);
    v16h r;
#pragma unroll
    for (int i = 0; i < 8; ++i) { r[i] = lo[i]; r[8 + i] = hi[i]; }
    return r;
}
// B-frag (32x16, lane->N=l%16): elem j -> K = 16h+j  => one contiguous 32B load
__device__ __forceinline__ v16h load_b_h(const _Float16* row, int h) {
    return *(const v16h*)(row + 16 * h);
}
__device__ __forceinline__ v16h load_a_f32(const float* row, int h) {
    v16h r;
#pragma unroll
    for (int i = 0; i < 8; ++i) {
        r[i]     = (_Float16)row[16 * h + i];
        r[8 + i] = (_Float16)row[16 * h + 8 + i];
    }
    return r;
}
__device__ __forceinline__ v16h load_b_f32(const float* row, int h) {
    const float* p = row + 16 * h;
    v16h r;
#pragma unroll
    for (int i = 0; i < 16; ++i) r[i] = (_Float16)p[i];
    return r;
}
// Combine two 16-row D-tiles (lo: rows 0-15, hi: rows 16-31) in D-layout
// (lane->col, reg r -> row 8h+r) into the B-fragment of that 32x16 block.
__device__ __forceinline__ v16h pack_b_from_pair(v8f lo, v8f hi, float scale, int h) {
    v16h r;
#pragma unroll
    for (int i = 0; i < 8; ++i) {
        float a  = lo[i] * scale;
        float b  = hi[i] * scale;
        float ax = __shfl_xor(a, 16, 32);
        float bx = __shfl_xor(b, 16, 32);
        r[i]     = (_Float16)(h ? bx : a);
        r[8 + i] = (_Float16)(h ? b : ax);
    }
    return r;
}

#if USE_TDM
// Issue a 2D TDM tile load: tile_d1 rows x tile_d0 f16 elems, row stride
// `stride_elems` (f16 units), packed contiguously at LDS byte offset lds_addr.
__device__ __forceinline__ void tdm_load_2d(unsigned lds_addr, const void* gptr,
                                            unsigned tile_d0, unsigned tile_d1,
                                            unsigned stride_elems) {
    unsigned long long ga = (unsigned long long)(uintptr_t)gptr;
    u32x4 g0;
    g0[0] = 1u;                                   // count=1, user descriptor
    g0[1] = lds_addr;                             // LDS byte address
    g0[2] = (unsigned)ga;                         // global_addr[31:0]
    g0[3] = (unsigned)(ga >> 32) | (2u << 30);    // global_addr[56:32] | type=2
    i32x8 g1;
    g1[0] = (int)(1u << 16);                      // wg_mask=0, data_size=1 (2B)
    g1[1] = (int)(tile_d0 << 16);                 // tensor_dim0[15:0] in [63:48]
    g1[2] = (int)(tile_d1 << 16);                 // tensor_dim1[15:0] in [95:80]
    g1[3] = (int)(tile_d0 << 16);                 // tile_dim0 in [127:112]
    g1[4] = (int)tile_d1;                         // tile_dim1 in [143:128]
    g1[5] = (int)stride_elems;                    // tensor_dim0_stride[31:0]
    g1[6] = 0;
    g1[7] = 0;
    i32x4 z4 = {};
#if __clang_major__ >= 23
    i32x8 z8 = {};
    __builtin_amdgcn_tensor_load_to_lds(g0, g1, z4, z4, z8, 0);
#else
    __builtin_amdgcn_tensor_load_to_lds(g0, g1, z4, z4, 0);
#endif
}
#endif

// ---------------- kernel 0: transpose weights to f16 ----------------
__global__ void k_prep(const float* __restrict__ wq, const float* __restrict__ wk,
                       const float* __restrict__ wv, const float* __restrict__ wp,
                       _Float16* __restrict__ wt) {
    int idx = blockIdx.x * blockDim.x + threadIdx.x;   // 0..65535
    int m = idx >> 14;
    int e = idx & 16383;
    int i = e >> 7, j = e & 127;
    const float* src = (m == 0) ? wq : (m == 1) ? wk : (m == 2) ? wv : wp;
    wt[m * 16384 + j * 128 + i] = (_Float16)src[i * 128 + j];
}

// ---------------- kernel 1: QKV projection (WMMA) ----------------
__global__ void k_qkv(const float* __restrict__ x, const float* __restrict__ bq,
                      const float* __restrict__ bk, const float* __restrict__ bv,
                      const _Float16* __restrict__ wqT, const _Float16* __restrict__ wkT,
                      const _Float16* __restrict__ wvT,
                      _Float16* __restrict__ Q, _Float16* __restrict__ K,
                      _Float16* __restrict__ Vt) {
    const int lane = threadIdx.x & 31;
    const int h = lane >> 4, lm = lane & 15;
    const int wave = blockIdx.x * (blockDim.x >> 5) + (threadIdx.x >> 5);
    const int b = wave >> 8;           // 256 tiles of 16 rows per batch
    const int m0 = (wave & 255) << 4;

    const float* xrow = x + (size_t)(b * Nn + m0 + lm) * Cc;
    v16h xa[4], xb[4];
#pragma unroll
    for (int c = 0; c < 4; ++c) {
        xa[c] = load_a_f32(xrow + 32 * c, h);
        xb[c] = load_b_f32(xrow + 32 * c, h);
    }

    const size_t qbase = (size_t)(b * Nn + m0 + lm) * HCc;
#pragma unroll 1
    for (int u = 0; u < 8; ++u) {
        v8f accq = {};
        v8f acck = {};
#pragma unroll
        for (int c = 0; c < 4; ++c) {
            v16h wfq = load_a_h(wqT + (16 * u + lm) * Cc + 32 * c, h);
            accq = WMMA_F16(wfq, xb[c], accq);
            v16h wfk = load_a_h(wkT + (16 * u + lm) * Cc + 32 * c, h);
            acck = WMMA_F16(wfk, xb[c], acck);
        }
        v8h qs, ks;
#pragma unroll
        for (int r = 0; r < 8; ++r) {
            qs[r] = (_Float16)(accq[r] + bq[16 * u + 8 * h + r]);
            ks[r] = (_Float16)(acck[r] + bk[16 * u + 8 * h + r]);
        }
        *(v8h*)(Q + qbase + 16 * u + 8 * h) = qs;
        *(v8h*)(K + qbase + 16 * u + 8 * h) = ks;

        v8f accv = {};
#pragma unroll
        for (int c = 0; c < 4; ++c) {
            v16h wfv = load_b_h(wvT + (16 * u + lm) * Cc + 32 * c, h);
            accv = WMMA_F16(xa[c], wfv, accv);
        }
        float bvv = bv[16 * u + lm];
        v8h vs;
#pragma unroll
        for (int r = 0; r < 8; ++r) vs[r] = (_Float16)(accv[r] + bvv);
        *(v8h*)(Vt + ((size_t)b * HCc + 16 * u + lm) * Nn + m0 + 8 * h) = vs;
    }
}

// ---------------- kernel 2: flash attention + fused projection ----------------
// Block = 8 waves = 128 query rows of one batch. K / V^T tiles for each 32-wide
// KV step are staged in LDS (TDM async DMA, double buffered) and shared by all
// 8 waves; each wave owns 16 query rows and keeps Q fragments + O^T in VGPRs.
__global__ void __launch_bounds__(256) k_attn(
        const float* __restrict__ x, const float* __restrict__ bp,
        const _Float16* __restrict__ Q, const _Float16* __restrict__ K,
        const _Float16* __restrict__ Vt, const _Float16* __restrict__ wpT,
        float* __restrict__ out) {
    const int lane = threadIdx.x & 31;
    const int h = lane >> 4, lm = lane & 15;
    const int wv_id = threadIdx.x >> 5;         // wave in block: 0..7
    const int b = blockIdx.x >> 5;              // 32 blocks per batch
    const int m0 = ((blockIdx.x & 31) << 7) + (wv_id << 4);

    __shared__ _Float16 shK[2][32 * 128];       // [buf][kv_local][d]
    __shared__ _Float16 shV[2][128 * 32];       // [buf][d][kv_local]

    const _Float16* Qb = Q + (size_t)b * Nn * HCc;
    const _Float16* Kb = K + (size_t)b * Nn * HCc;
    const _Float16* Vb = Vt + (size_t)b * HCc * Nn;

    v16h qb[4];  // resident B-fragments of Q^T (lane -> query row)
#pragma unroll
    for (int c = 0; c < 4; ++c) qb[c] = load_b_h(Qb + (m0 + lm) * HCc + 32 * c, h);

    v8f o[8];  // O^T accumulators: lane->m, tile t covers d = 16t + 8h + r
#pragma unroll
    for (int t = 0; t < 8; ++t)
#pragma unroll
        for (int r = 0; r < 8; ++r) o[t][r] = 0.0f;
    float m_run = -3.0e38f, l_run = 0.0f;

#if USE_TDM
    if (wv_id == 0) {
        tdm_load_2d((unsigned)(uintptr_t)&shK[0][0], Kb, 128u, 32u, (unsigned)HCc);
        tdm_load_2d((unsigned)(uintptr_t)&shV[0][0], Vb, 32u, 128u, (unsigned)Nn);
    }
#endif

    for (int it = 0; it < Nn / 32; ++it) {
        const int kv0 = it * 32;
#if USE_TDM
        const int buf = it & 1;
        if (wv_id == 0) __builtin_amdgcn_s_wait_tensorcnt(0);
        __syncthreads();  // staged tiles visible to all waves
        if (wv_id == 0 && it + 1 < Nn / 32) {
            tdm_load_2d((unsigned)(uintptr_t)&shK[buf ^ 1][0],
                        Kb + (size_t)(kv0 + 32) * HCc, 128u, 32u, (unsigned)HCc);
            tdm_load_2d((unsigned)(uintptr_t)&shV[buf ^ 1][0],
                        Vb + (kv0 + 32), 32u, 128u, (unsigned)Nn);
        }
#else
        const int buf = 0;
        {   // cooperative staging: K tile is one contiguous 8KB block
            const v8h* srcK = (const v8h*)(Kb + (size_t)kv0 * HCc);
            v8h* dstK = (v8h*)&shK[0][0];
            v8h* dstV = (v8h*)&shV[0][0];
#pragma unroll
            for (int i = 0; i < 2; ++i) {
                int idx = threadIdx.x + 256 * i;
                dstK[idx] = srcK[idx];
                int rowd = idx >> 2, cchunk = idx & 3;
                dstV[idx] = *(const v8h*)(Vb + (size_t)rowd * Nn + kv0 + 8 * cchunk);
            }
        }
        __syncthreads();
#endif
        const _Float16* kt = &shK[buf][0];
        const _Float16* vt = &shV[buf][0];

        // S^T = K_tile @ Q^T  (two 16x16 D tiles covering kv0..kv0+31)
        v8f s0 = {};
        v8f s1 = {};
#pragma unroll
        for (int c = 0; c < 4; ++c) {
            v16h ka = load_a_h(kt + lm * 128 + 32 * c, h);
            s0 = WMMA_F16(ka, qb[c], s0);
        }
#pragma unroll
        for (int c = 0; c < 4; ++c) {
            v16h ka = load_a_h(kt + (16 + lm) * 128 + 32 * c, h);
            s1 = WMMA_F16(ka, qb[c], s1);
        }
        // online softmax: lane owns 16 of the 32 scores of row m; lane^16 the rest
        float tmax = s0[0];
#pragma unroll
        for (int r = 0; r < 8; ++r) {
            tmax = fmaxf(tmax, s0[r]);
            tmax = fmaxf(tmax, s1[r]);
        }
        tmax = fmaxf(tmax, __shfl_xor(tmax, 16, 32));
        float mnew = fmaxf(m_run, tmax);
        float alpha = __expf(m_run - mnew);
        v8f p0, p1;
        float lsum = 0.0f;
#pragma unroll
        for (int r = 0; r < 8; ++r) {
            p0[r] = __expf(s0[r] - mnew);
            p1[r] = __expf(s1[r] - mnew);
            lsum += p0[r] + p1[r];
        }
        lsum += __shfl_xor(lsum, 16, 32);
        l_run = l_run * alpha + lsum;
        m_run = mnew;
#pragma unroll
        for (int t = 0; t < 8; ++t)
#pragma unroll
            for (int r = 0; r < 8; ++r) o[t][r] *= alpha;

        // O^T += V^T @ P^T : P^T B-frag comes straight from the S^T tiles
        v16h pB = pack_b_from_pair(p0, p1, 1.0f, h);
#pragma unroll
        for (int t = 0; t < 8; ++t) {
            v16h va = load_a_h(vt + (16 * t + lm) * 32, h);
            o[t] = WMMA_F16(va, pB, o[t]);
        }
        __syncthreads();  // all waves done with this buffer
    }

    // proj^T = wp^T @ (O^T / l) ; normalization folded into fragment repack
    float inv_l = 1.0f / l_run;
    v16h ofB[4];
#pragma unroll
    for (int c = 0; c < 4; ++c) ofB[c] = pack_b_from_pair(o[2 * c], o[2 * c + 1], inv_l, h);

    const int row = m0 + lm;  // global query index for this lane
    const float* xr = x + (size_t)(b * Nn + row) * Cc;
    float* orow = out + (size_t)(b * Nn + row) * Cc;
#pragma unroll 1
    for (int u = 0; u < 8; ++u) {
        v8f pa = {};
#pragma unroll
        for (int c = 0; c < 4; ++c) {
            v16h wf = load_a_h(wpT + (16 * u + lm) * HCc + 32 * c, h);
            pa = WMMA_F16(wf, ofB[c], pa);
        }
        const int ch = 16 * u + 8 * h;  // pa reg r -> out channel ch + r
        v4f x0 = *(const v4f*)(xr + ch);
        v4f x1 = *(const v4f*)(xr + ch + 4);
        v4f b0 = *(const v4f*)(bp + ch);
        v4f b1 = *(const v4f*)(bp + ch + 4);
        v4f r0, r1;
#pragma unroll
        for (int r = 0; r < 4; ++r) {
            r0[r] = fmaxf(x0[r] + pa[r] + b0[r], 0.0f);
            r1[r] = fmaxf(x1[r] + pa[4 + r] + b1[r], 0.0f);
        }
        *(v4f*)(orow + ch) = r0;
        *(v4f*)(orow + ch + 4) = r1;
    }
}

extern "C" void kernel_launch(void* const* d_in, const int* in_sizes, int n_in,
                              void* d_out, int out_size, void* d_ws, size_t ws_size,
                              hipStream_t stream) {
    const float* x  = (const float*)d_in[0];
    const float* wq = (const float*)d_in[1];
    const float* bq = (const float*)d_in[2];
    const float* wk = (const float*)d_in[3];
    const float* bk = (const float*)d_in[4];
    const float* wv = (const float*)d_in[5];
    const float* bv = (const float*)d_in[6];
    const float* wp = (const float*)d_in[7];
    const float* bp = (const float*)d_in[8];
    float* out = (float*)d_out;

    _Float16* ws16 = (_Float16*)d_ws;
    _Float16* wqT = ws16;                    // 4 x 128x128 f16 transposed weights
    _Float16* wkT = ws16 + 16384;
    _Float16* wvT = ws16 + 32768;
    _Float16* wpT = ws16 + 49152;
    _Float16* Qm = ws16 + 65536;             // [B][N][HC] f16
    _Float16* Km = Qm + (size_t)Bn * Nn * HCc;
    _Float16* Vt = Km + (size_t)Bn * Nn * HCc;  // [B][HC][N] f16 (transposed)
    (void)in_sizes; (void)n_in; (void)out_size; (void)ws_size;

    k_prep<<<256, 256, 0, stream>>>(wq, wk, wv, wp, ws16);
    k_qkv<<<256, 256, 0, stream>>>(x, bq, bk, bv, wqT, wkT, wvT, Qm, Km, Vt);
    k_attn<<<256, 256, 0, stream>>>(x, bp, Qm, Km, Vt, wpT, out);
}
